// cheb_conv_withSAt_39273180955556
// MI455X (gfx1250) — matrix operand order repaired and checked by reading, not compiled, not run on hardware
//
#include <hip/hip_runtime.h>
#include <stdint.h>

// Problem constants (from reference): B=32, K=3, N=512, F_IN=F_OUT=64, T=64
#define B_ 32
#define K_ 3
#define N_ 512
#define F_ 64
#define T_ 64
#define C_ (F_ * T_)   // 4096 fused (o,t) columns
#define CH 4           // batch chunk (workspace ~74 MB)

typedef __attribute__((ext_vector_type(16))) __bf16 bf16x16;
typedef __attribute__((ext_vector_type(8)))  float  v8f;

union Frag { bf16x16 v; uint4 u[2]; };

__device__ __forceinline__ unsigned short f2bf(float f) {
  unsigned int u = __float_as_uint(f);
  u += 0x7FFFu + ((u >> 16) & 1u);   // round-to-nearest-even
  return (unsigned short)(u >> 16);
}

// WMMA A-operand (16x32 bf16), row-major storage: lane L<16 holds row, K 0..7 & 16..23;
// lane L>=16 holds K 8..15 & 24..31  ->  two b128 loads per lane.
__device__ __forceinline__ bf16x16 load_frag_rowmajor(const char* base, int row,
                                                      int rowStrideBytes, int kByteOff,
                                                      int lane) {
  Frag f;
  const char* p = base + (size_t)row * rowStrideBytes + kByteOff + ((lane & 16) ? 16 : 0);
  f.u[0] = *reinterpret_cast<const uint4*>(p);
  f.u[1] = *reinterpret_cast<const uint4*>(p + 32);
  return f.v;
}

// WMMA B-operand (32x16 bf16), column-contiguous-over-K storage: lane L<16 holds K 0..7 & 16..23
// of column L; lane L>=16 holds K 8..15 & 24..31  ->  two b128 loads per lane.
__device__ __forceinline__ bf16x16 load_frag_colmajor(const char* base, int col,
                                                      int colStrideBytes, int kByteOff,
                                                      int lane) {
  Frag f;
  const char* p = base + (size_t)col * colStrideBytes + kByteOff + ((lane & 16) ? 16 : 0);
  f.u[0] = *reinterpret_cast<const uint4*>(p);
  f.u[1] = *reinterpret_cast<const uint4*>(p + 32);
  return f.v;
}

__device__ __forceinline__ v8f wmma_bf16(bf16x16 a, bf16x16 b, v8f c) {
  return __builtin_amdgcn_wmma_f32_16x16x32_bf16(false, a, false, b, (short)0, c,
                                                 false, false);
}

// ---------------------------------------------------------------------------
// Kernel 1: fused (adj*mask + scores) -> rowwise softmax -> *cheb -> bf16
// One wave32 per row of 512 neighbors; 8 rows per 256-thread block.
// ---------------------------------------------------------------------------
__global__ void softmax_aeff_kernel(const float* __restrict__ scores,
                                    const float* __restrict__ adj,
                                    const float* __restrict__ mask,
                                    const float* __restrict__ cheb,
                                    unsigned short* __restrict__ Abf, int b0) {
  int lane = threadIdx.x & 31;
  int wv   = threadIdx.x >> 5;
  int row  = blockIdx.x * 8 + wv;          // [0, CH*K_*N_)
  int brel = row / (K_ * N_);
  int k    = (row / N_) % K_;
  int n    = row % N_;
  int b    = b0 + brel;

  const float* sc = scores + (((size_t)(b * K_ + k) * N_ + n) * N_);
  const float* ad = adj + (size_t)n * N_;
  const float* mk = mask + ((size_t)(k * N_ + n) * N_);
  const float* cb = cheb + ((size_t)(k * N_ + n) * N_);
  unsigned short* outp = Abf + (((size_t)(brel * K_ + k) * N_ + n) * N_);

  float v[16];
  float mx = -1e30f;
#pragma unroll
  for (int j = 0; j < 16; ++j) {
    int m = lane + 32 * j;
    float c = sc[m] + ad[m] * mk[m];
    v[j] = c;
    mx = fmaxf(mx, c);
  }
#pragma unroll
  for (int o = 16; o > 0; o >>= 1) mx = fmaxf(mx, __shfl_xor(mx, o, 32));
  float s = 0.f;
#pragma unroll
  for (int j = 0; j < 16; ++j) { v[j] = __expf(v[j] - mx); s += v[j]; }
#pragma unroll
  for (int o = 16; o > 0; o >>= 1) s += __shfl_xor(s, o, 32);
  float inv = __frcp_rn(s);
#pragma unroll
  for (int j = 0; j < 16; ++j) {
    int m = lane + 32 * j;
    outp[m] = f2bf(v[j] * inv * cb[m]);
  }
}

// ---------------------------------------------------------------------------
// Kernel 2: Theta [k][i][o] fp32 -> ThetaT [k][o][i] bf16 (WMMA A-operand)
// ---------------------------------------------------------------------------
__global__ void thetaT_kernel(const float* __restrict__ Theta,
                              unsigned short* __restrict__ thT) {
  int gid = blockIdx.x * 256 + threadIdx.x;  // < K_*F_*F_ = 12288
  int k = gid >> 12;
  int r = gid & 4095;
  int o = r >> 6;
  int i = r & 63;
  thT[gid] = f2bf(Theta[((size_t)k * F_ + i) * F_ + o]);
  (void)o;
}

// ---------------------------------------------------------------------------
// Kernel 3: x[b][m][i][t] fp32 -> Xp[brel][t][m][i] bf16 via LDS transpose.
// One block per (brel, m): 64x64 tile.
// ---------------------------------------------------------------------------
__global__ void xprep_kernel(const float* __restrict__ x,
                             unsigned short* __restrict__ Xp, int b0) {
  __shared__ unsigned short lds[64][65];
  int brel = blockIdx.x / N_;
  int m    = blockIdx.x % N_;
  int b    = b0 + brel;
  const float* src = x + ((size_t)(b * N_ + m)) * (F_ * T_);
  int tid = threadIdx.x;
#pragma unroll
  for (int j = 0; j < 16; ++j) {
    int idx = tid + j * 256;                 // idx = i*64 + t (coalesced)
    lds[idx >> 6][idx & 63] = f2bf(src[idx]);
  }
  __syncthreads();
#pragma unroll
  for (int j = 0; j < 16; ++j) {
    int idx = tid + j * 256;                 // idx = t*64 + i
    int t = idx >> 6, i = idx & 63;
    Xp[(((size_t)(brel * T_ + t) * N_ + m) * F_) + i] = lds[i][t];
  }
}

// ---------------------------------------------------------------------------
// Kernel 4: Z[brel][k][(o*64+t)][m] = sum_i ThetaT[k][o][i] * Xp[brel][t][m][i]
// Per block: (brel, k, t, m-half). 8 waves; each wave: 2 o-tiles x 4 m-tiles,
// K = F_IN = 64 -> 2 WMMA steps.  16 v_wmma per wave.
// launch_bounds(256,1): full VGPR budget, no spills.
// ---------------------------------------------------------------------------
__global__ void __launch_bounds__(256, 1)
zcompute_kernel(const unsigned short* __restrict__ thT,
                const unsigned short* __restrict__ Xp,
                unsigned short* __restrict__ Z) {
  int lane = threadIdx.x & 31;
  int lanelo = lane & 15;
  int wv = threadIdx.x >> 5;
  int w0 = wv >> 2;   // o half (0..1)
  int w1 = wv & 3;    // m group (0..3)

  int bid  = blockIdx.x;
  int brel = bid / (K_ * T_ * 2);
  int rem  = bid % (K_ * T_ * 2);
  int k    = rem / (T_ * 2);
  int r2   = rem % (T_ * 2);
  int t    = r2 >> 1;
  int mh   = r2 & 1;

  const char* thbase = (const char*)(thT + (size_t)k * F_ * F_);
  const char* xbase  = (const char*)(Xp + ((size_t)(brel * T_ + t) * N_) * F_);

  const v8f zero = {0.f, 0.f, 0.f, 0.f, 0.f, 0.f, 0.f, 0.f};
  v8f acc[2][4];
#pragma unroll
  for (int a = 0; a < 2; ++a)
#pragma unroll
    for (int mt = 0; mt < 4; ++mt) acc[a][mt] = zero;

#pragma unroll
  for (int ks = 0; ks < 2; ++ks) {
    int i0 = ks * 32;
    bf16x16 af[2], bfr[4];
#pragma unroll
    for (int a = 0; a < 2; ++a)
      af[a] = load_frag_rowmajor(thbase, (2 * w0 + a) * 16 + lanelo, F_ * 2, i0 * 2, lane);
#pragma unroll
    for (int mt = 0; mt < 4; ++mt) {
      int m = mh * 256 + w1 * 64 + mt * 16 + lanelo;
      bfr[mt] = load_frag_colmajor(xbase, m, F_ * 2, i0 * 2, lane);
    }
#pragma unroll
    for (int a = 0; a < 2; ++a)
#pragma unroll
      for (int mt = 0; mt < 4; ++mt) acc[a][mt] = wmma_bf16(af[a], bfr[mt], acc[a][mt]);
  }

  unsigned short* zb = Z + ((size_t)(brel * K_ + k) * C_) * N_;
#pragma unroll
  for (int a = 0; a < 2; ++a) {
    int ob = w0 * 32 + a * 16 + ((lane & 16) ? 8 : 0);
#pragma unroll
    for (int mt = 0; mt < 4; ++mt) {
      int m = mh * 256 + w1 * 64 + mt * 16 + lanelo;
#pragma unroll
      for (int r = 0; r < 8; ++r) {
        int o = ob + r;
        zb[((size_t)(o * T_ + t)) * N_ + m] = f2bf(acc[a][mt][r]);
      }
    }
  }
}

// ---------------------------------------------------------------------------
// Kernel 5: Out[b][n][(o,t)] = relu( sum_{k,m} Abf[brel][k][n][m]*Z[brel][k][c][m] )
// K-sum over Chebyshev orders folded into GEMM-K (3*512 = 1536).
// Block tile 64n x 256c; wave tile 32n x 64c (2x4 accum tiles):
// 8 v_wmma per 6 fragment loads per K-step; 384 v_wmma per wave.
// ---------------------------------------------------------------------------
__global__ void __launch_bounds__(256, 1)
gemm_out_kernel(const unsigned short* __restrict__ Abf,
                const unsigned short* __restrict__ Z,
                float* __restrict__ out, int b0) {
  int lane = threadIdx.x & 31;
  int lanelo = lane & 15;
  int wv = threadIdx.x >> 5;
  int w0 = wv >> 2;   // n half (0..1)
  int w1 = wv & 3;    // c quarter (0..3)

  int bid  = blockIdx.x;
  int brel = bid / (8 * 16);
  int rem  = bid % (8 * 16);
  int nt   = rem / 16;        // n block tile (64 rows)
  int ct   = rem % 16;        // c block tile (256 cols)

  int nb = nt * 64 + w0 * 32;
  int cb = ct * 256 + w1 * 64;

  const v8f zero = {0.f, 0.f, 0.f, 0.f, 0.f, 0.f, 0.f, 0.f};
  v8f acc[2][4];
#pragma unroll
  for (int i = 0; i < 2; ++i)
#pragma unroll
    for (int j = 0; j < 4; ++j) acc[i][j] = zero;

#pragma unroll 1
  for (int k = 0; k < K_; ++k) {
    const char* abase = (const char*)(Abf + ((size_t)(brel * K_ + k) * N_) * N_);
    const char* zbase = (const char*)(Z + ((size_t)(brel * K_ + k) * C_) * N_);
#pragma unroll 1
    for (int mt = 0; mt < 16; ++mt) {
      int m0 = mt * 32;
      if (mt < 15) {   // speculative prefetch of the next K-slice (global_prefetch_b8)
        __builtin_prefetch(abase + (size_t)(nb + lanelo) * (N_ * 2) + (m0 + 32) * 2, 0, 0);
        __builtin_prefetch(zbase + (size_t)(cb + lanelo) * (N_ * 2) + (m0 + 32) * 2, 0, 0);
      }
      bf16x16 af[2], bfr[4];
#pragma unroll
      for (int i = 0; i < 2; ++i)
        af[i] = load_frag_rowmajor(abase, nb + i * 16 + lanelo, N_ * 2, m0 * 2, lane);
#pragma unroll
      for (int j = 0; j < 4; ++j)
        bfr[j] = load_frag_colmajor(zbase, cb + j * 16 + lanelo, N_ * 2, m0 * 2, lane);
#pragma unroll
      for (int i = 0; i < 2; ++i)
#pragma unroll
        for (int j = 0; j < 4; ++j) acc[i][j] = wmma_bf16(af[i], bfr[j], acc[i][j]);
    }
  }

  int b = b0 + brel;
  float* ob = out + ((size_t)b * N_) * C_;
#pragma unroll
  for (int i = 0; i < 2; ++i) {
    int nrow = nb + i * 16 + ((lane & 16) ? 8 : 0);
#pragma unroll
    for (int j = 0; j < 4; ++j) {
      int c = cb + j * 16 + lanelo;
#pragma unroll
      for (int r = 0; r < 8; ++r) {
        ob[((size_t)(nrow + r)) * C_ + c] = fmaxf(acc[i][j][r], 0.f);  // fused ReLU
      }
    }
  }
}

// ---------------------------------------------------------------------------
extern "C" void kernel_launch(void* const* d_in, const int* in_sizes, int n_in,
                              void* d_out, int out_size, void* d_ws, size_t ws_size,
                              hipStream_t stream) {
  const float* x      = (const float*)d_in[0];
  const float* scores = (const float*)d_in[1];
  const float* adj    = (const float*)d_in[2];
  const float* cheb   = (const float*)d_in[3];
  const float* Theta  = (const float*)d_in[4];
  const float* mask   = (const float*)d_in[5];
  float* out = (float*)d_out;

  char* ws = (char*)d_ws;
  size_t szA  = (size_t)CH * K_ * N_ * N_ * 2;         //  6,291,456 B
  size_t szTh = (((size_t)K_ * F_ * F_ * 2) + 255) & ~(size_t)255;  // 24,576 B
  size_t szXp = (size_t)CH * T_ * N_ * F_ * 2;         // 16,777,216 B
  // Z: 50,331,648 B  -> total ~74 MB of d_ws
  unsigned short* Abf = (unsigned short*)(ws);
  unsigned short* thT = (unsigned short*)(ws + szA);
  unsigned short* Xp  = (unsigned short*)(ws + szA + szTh);
  unsigned short* Z   = (unsigned short*)(ws + szA + szTh + szXp);

  thetaT_kernel<<<(K_ * F_ * F_) / 256, 256, 0, stream>>>(Theta, thT);

  for (int c = 0; c < B_ / CH; ++c) {
    int b0 = c * CH;
    softmax_aeff_kernel<<<(CH * K_ * N_) / 8, 256, 0, stream>>>(scores, adj, mask, cheb,
                                                                Abf, b0);
    xprep_kernel<<<CH * N_, 256, 0, stream>>>(x, Xp, b0);
    zcompute_kernel<<<CH * K_ * T_ * 2, 256, 0, stream>>>(thT, Xp, Z);
    gemm_out_kernel<<<CH * 8 * 16, 256, 0, stream>>>(Abf, Z, out, b0);
  }
}
